// SimpleDecoder_43937515438313
// MI455X (gfx1250) — matrix operand order repaired and checked by reading, not compiled, not run on hardware
//
#include <hip/hip_runtime.h>
#include <hip/hip_bf16.h>

#define S   512
#define G   1536
#define BB  32
#define TT  128
#define BT  4096
#define V   32000

typedef __attribute__((ext_vector_type(16))) __bf16 v16bf;
typedef __attribute__((ext_vector_type(8)))  float  v8f;
typedef __attribute__((ext_vector_type(4)))  int    v4i;

struct P2 { uint4 lo, hi; };

#if __has_builtin(__builtin_amdgcn_global_load_async_to_lds_b128) && \
    __has_builtin(__builtin_amdgcn_s_wait_asynccnt)
#define HAVE_ASYNC_LDS 1
#else
#define HAVE_ASYNC_LDS 0
#endif

__device__ __forceinline__ unsigned short f2bf(float f) {
  unsigned u = __builtin_bit_cast(unsigned, f);
  u += 0x7FFFu + ((u >> 16) & 1u);          // round-to-nearest-even
  return (unsigned short)(u >> 16);
}

// A fragment: 16x32 bf16 tile from row-major [M,K]; lanes0-15 hold K=0..7 /
// 16..23, lanes16-31 hold K=8..15 / 24..31 (ISA 16-bit A layout).
__device__ __forceinline__ v16bf fragA(const unsigned short* tile, int ldk, int lane) {
  int r    = lane & 15;
  int koff = (lane & 16) ? 8 : 0;
  const unsigned short* p = tile + (size_t)r * ldk + koff;
  P2 t; t.lo = *(const uint4*)p; t.hi = *(const uint4*)(p + 16);
  return __builtin_bit_cast(v16bf, t);
}
// B fragment: 32x16 tile, B = W^T with W row-major [N,K]; lane n<16 holds
// K=0..7 / 16..23, lane n+16 holds K=8..15 / 24..31 (matches A layout).
__device__ __forceinline__ v16bf fragB(const unsigned short* tile, int ldk, int lane) {
  int n    = lane & 15;
  int koff = (lane & 16) ? 8 : 0;
  const unsigned short* p = tile + (size_t)n * ldk + koff;
  P2 t; t.lo = *(const uint4*)p; t.hi = *(const uint4*)(p + 16);
  return __builtin_bit_cast(v16bf, t);
}

// 16B global -> LDS copy, async (ASYNCcnt) when available.
__device__ __forceinline__ void cp16(unsigned short* lds_dst, const unsigned short* g_src) {
#if HAVE_ASYNC_LDS
  __builtin_amdgcn_global_load_async_to_lds_b128((v4i*)g_src, (v4i*)lds_dst, 0, 0);
#else
  *(uint4*)lds_dst = *(const uint4*)g_src;
#endif
}
__device__ __forceinline__ void wait_async_le5() {
#if HAVE_ASYNC_LDS
  __builtin_amdgcn_s_wait_asynccnt(5);
#endif
}
__device__ __forceinline__ void wait_async_le0() {
#if HAVE_ASYNC_LDS
  __builtin_amdgcn_s_wait_asynccnt(0);
#endif
}

// ---------------- init: h0 concat, zero weights tail, zero barrier ----------
__global__ void k_init(const float* __restrict__ lh, float* __restrict__ h32,
                       unsigned short* __restrict__ hb, float* __restrict__ out_tail,
                       unsigned* __restrict__ bar) {
  int i = blockIdx.x * blockDim.x + threadIdx.x;
  if (i < BB * S) {
    int b = i >> 9, j = i & 511;
    float v = (j < 256) ? lh[b * 256 + j] : lh[BB * 256 + b * 256 + (j - 256)];
    h32[i] = v;
    hb[i]  = f2bf(v);
  }
  if (i < 3200) out_tail[i] = 0.f;   // weights output = zeros
  if (i < 2)    bar[i] = 0u;         // grid-barrier {count, gen}
}

// ---------------- fp32 -> bf16 weight conversion ----------------------------
__global__ void k_cvt(const float* __restrict__ src, unsigned short* __restrict__ dst, int n) {
  int i  = blockIdx.x * blockDim.x + threadIdx.x;
  int st = gridDim.x * blockDim.x;
  for (; i < n; i += st) dst[i] = f2bf(src[i]);
}

// ---------------- embedding gather + bf16 convert ---------------------------
__global__ void k_gather(const int* __restrict__ inp, const float* __restrict__ emb,
                         unsigned short* __restrict__ xb) {
  int i  = blockIdx.x * blockDim.x + threadIdx.x;
  int st = gridDim.x * blockDim.x;
  for (; i < BT * S; i += st) {
    int bt = i >> 9, s = i & 511;
    xb[i] = f2bf(emb[(size_t)inp[bt] * S + s]);
  }
}

// ---------------- gi = x @ w_ih^T + b_ih  (WMMA bf16, fp32 accum) -----------
__global__ void __launch_bounds__(256) k_gi(const unsigned short* __restrict__ xb,
                                            const unsigned short* __restrict__ wib,
                                            const float* __restrict__ b_ih,
                                            float* __restrict__ gi) {
  int lane = threadIdx.x & 31, wid = threadIdx.x >> 5;
  int m0 = blockIdx.x * 128 + wid * 16;
  int n0 = blockIdx.y * 64;
  v8f acc[4] = {};
  for (int k0 = 0; k0 < S; k0 += 32) {
    v16bf a = fragA(xb + (size_t)m0 * S + k0, S, lane);
#pragma unroll
    for (int j = 0; j < 4; ++j) {
      v16bf b = fragB(wib + (size_t)(n0 + j * 16) * S + k0, S, lane);
      acc[j] = __builtin_amdgcn_wmma_f32_16x16x32_bf16(false, a, false, b, (short)0, acc[j], false, false);
    }
  }
  int n_l = lane & 15, mo = (lane & 16) ? 8 : 0;
#pragma unroll
  for (int j = 0; j < 4; ++j) {
    float bias = b_ih[n0 + j * 16 + n_l];
#pragma unroll
    for (int v = 0; v < 8; ++v)
      gi[(size_t)(m0 + v + mo) * G + n0 + j * 16 + n_l] = acc[j][v] + bias;
  }
}

// ---------------- persistent GRU: 32 blocks, grid barrier per step ----------
__global__ void __launch_bounds__(192) k_gru(const float* __restrict__ w_hh,
                                             const float* __restrict__ b_hh,
                                             const float* __restrict__ gi,
                                             float* __restrict__ h32,
                                             unsigned short* __restrict__ hb,
                                             unsigned short* __restrict__ hs,
                                             unsigned* __restrict__ bar) {
  __shared__ __align__(16) unsigned short wlds[48 * S];
  __shared__ __align__(16) float ghs[3][32][16];
  const int tid = threadIdx.x;
  const int lane = tid & 31, wid = tid >> 5;
  const int j0 = blockIdx.x * 16;

  for (int i = tid; i < 48 * S; i += 192) {
    int lr = i >> 9, k = i & 511;
    int g = lr >> 4, jj = lr & 15;
    wlds[i] = f2bf(w_hh[(size_t)(g * 512 + j0 + jj) * S + k]);
  }
  __syncthreads();

  const int mhalf = wid & 1, gate = wid >> 1;
  int cur = 0;
  for (int t = 0; t < TT; ++t) {
    const unsigned short* hcur = hb + cur * BB * S;
    v8f acc = {};
    for (int k0 = 0; k0 < S; k0 += 32) {
      v16bf a = fragA(hcur + (size_t)mhalf * 16 * S + k0, S, lane);
      v16bf b = fragB(wlds + (size_t)gate * 16 * S + k0, S, lane);
      acc = __builtin_amdgcn_wmma_f32_16x16x32_bf16(false, a, false, b, (short)0, acc, false, false);
    }
    {
      int n_l = lane & 15, mo = (lane & 16) ? 8 : 0;
#pragma unroll
      for (int v = 0; v < 8; ++v)
        ghs[gate][mhalf * 16 + v + mo][n_l] = acc[v];
    }
    __syncthreads();

    int nxt = cur ^ 1;
    for (int e = tid; e < BB * 16; e += 192) {
      int b = e >> 4, jj = e & 15, j = j0 + jj;
      size_t gir = (size_t)(b * TT + t) * G;
      float ir  = gi[gir + j];
      float iz  = gi[gir + 512 + j];
      float in_ = gi[gir + 1024 + j];
      float hr = ghs[0][b][jj] + b_hh[j];
      float hz = ghs[1][b][jj] + b_hh[512 + j];
      float hn = ghs[2][b][jj] + b_hh[1024 + j];
      float r = 1.f / (1.f + expf(-(ir + hr)));
      float z = 1.f / (1.f + expf(-(iz + hz)));
      float n = tanhf(in_ + r * hn);
      float hp = h32[cur * BB * S + b * S + j];
      float hw = (1.f - z) * n + z * hp;
      h32[nxt * BB * S + b * S + j] = hw;
      unsigned short hwb = f2bf(hw);
      hb[nxt * BB * S + b * S + j] = hwb;
      hs[(size_t)(b * TT + t) * S + j] = hwb;
    }

    __syncthreads();
    if (tid == 0) {
      unsigned g0 = __hip_atomic_load(&bar[1], __ATOMIC_ACQUIRE, __HIP_MEMORY_SCOPE_AGENT);
      __threadfence();
      unsigned a = __hip_atomic_fetch_add(&bar[0], 1u, __ATOMIC_ACQ_REL, __HIP_MEMORY_SCOPE_AGENT);
      if (a == 31u) {
        __hip_atomic_store(&bar[0], 0u, __ATOMIC_RELAXED, __HIP_MEMORY_SCOPE_AGENT);
        __hip_atomic_fetch_add(&bar[1], 1u, __ATOMIC_RELEASE, __HIP_MEMORY_SCOPE_AGENT);
      } else {
        while (__hip_atomic_load(&bar[1], __ATOMIC_ACQUIRE, __HIP_MEMORY_SCOPE_AGENT) == g0)
          __builtin_amdgcn_s_sleep(1);
      }
    }
    __syncthreads();
    cur = nxt;
  }
}

// ---------------- logits = hs @ w_out^T + b_out -----------------------------
// Double-buffered async LDS staging: block tile M=64, N=256, K-step 32.
// 8 waves as 2x4; wave computes 2 M-tiles x 4 N-tiles (8 wmma / k-step).
__global__ void __launch_bounds__(256) k_proj(const unsigned short* __restrict__ hs,
                                              const unsigned short* __restrict__ wob,
                                              const float* __restrict__ b_out,
                                              float* __restrict__ out) {
  __shared__ __align__(16) unsigned short Als[2][64 * 32];    //  8 KB
  __shared__ __align__(16) unsigned short Bls[2][256 * 32];   // 32 KB
  const int tid  = threadIdx.x;
  const int lane = tid & 31, wid = tid >> 5;
  const int wm = wid & 1, wn = wid >> 1;
  const int Mblk = blockIdx.x * 64;
  const int Nblk = blockIdx.y * 256;

  // stage copier: 5 x 16B chunks per thread (1 A + 4 B)
  auto stage = [&](int ks, int buf) {
    const int k0 = ks * 32;
    {
      int row = tid >> 2, c = tid & 3;                         // 64 rows x 4 chunks
      cp16(&Als[buf][row * 32 + c * 8],
           hs + (size_t)(Mblk + row) * S + k0 + c * 8);
    }
#pragma unroll
    for (int q = 0; q < 4; ++q) {                              // 256 rows x 4 chunks
      int idx = tid + q * 256;
      int row = idx >> 2, c = idx & 3;
      cp16(&Bls[buf][row * 32 + c * 8],
           wob + (size_t)(Nblk + row) * S + k0 + c * 8);
    }
  };

  v8f acc[2][4] = {};
  stage(0, 0);
  for (int ks = 0; ks < 16; ++ks) {
    const int cur = ks & 1;
    if (ks < 15) { stage(ks + 1, cur ^ 1); wait_async_le5(); }
    else         { wait_async_le0(); }
    __syncthreads();

    v16bf a0 = fragA(&Als[cur][(wm * 32 + 0)  * 32], 32, lane);
    v16bf a1 = fragA(&Als[cur][(wm * 32 + 16) * 32], 32, lane);
#pragma unroll
    for (int j = 0; j < 4; ++j) {
      v16bf b = fragB(&Bls[cur][(wn * 64 + j * 16) * 32], 32, lane);
      acc[0][j] = __builtin_amdgcn_wmma_f32_16x16x32_bf16(false, a0, false, b, (short)0, acc[0][j], false, false);
      acc[1][j] = __builtin_amdgcn_wmma_f32_16x16x32_bf16(false, a1, false, b, (short)0, acc[1][j], false, false);
    }
    __syncthreads();   // all waves done reading buf[cur] before it is re-staged
  }

  const int m0 = Mblk + wm * 32;
  const int n0 = Nblk + wn * 64;
  int n_l = lane & 15, mo = (lane & 16) ? 8 : 0;
#pragma unroll
  for (int j = 0; j < 4; ++j) {
    float bias = b_out[n0 + j * 16 + n_l];
#pragma unroll
    for (int i = 0; i < 2; ++i)
#pragma unroll
      for (int v = 0; v < 8; ++v)
        out[(size_t)(m0 + i * 16 + v + mo) * V + n0 + j * 16 + n_l] = acc[i][j][v] + bias;
  }
}

// ---------------- workspace layout (bytes, all 256-aligned) -----------------
static constexpr size_t OFF_GI  = 0;
static constexpr size_t OFF_HSB = OFF_GI  + (size_t)BT * G * 4;
static constexpr size_t OFF_XB  = OFF_HSB + (size_t)BT * S * 2;
static constexpr size_t OFF_WOB = OFF_XB  + (size_t)BT * S * 2;
static constexpr size_t OFF_WIB = OFF_WOB + (size_t)V  * S * 2;
static constexpr size_t OFF_H32 = OFF_WIB + (size_t)G  * S * 2;
static constexpr size_t OFF_HB  = OFF_H32 + (size_t)2 * BB * S * 4;
static constexpr size_t OFF_BAR = OFF_HB  + (size_t)2 * BB * S * 2;

extern "C" void kernel_launch(void* const* d_in, const int* in_sizes, int n_in,
                              void* d_out, int out_size, void* d_ws, size_t ws_size,
                              hipStream_t stream) {
  const int*   inp   = (const int*)d_in[0];
  const float* lh    = (const float*)d_in[2];
  const float* emb   = (const float*)d_in[3];
  const float* w_ih  = (const float*)d_in[4];
  const float* w_hh  = (const float*)d_in[5];
  const float* b_ih  = (const float*)d_in[6];
  const float* b_hh  = (const float*)d_in[7];
  const float* w_out = (const float*)d_in[8];
  const float* b_out = (const float*)d_in[9];
  float* out = (float*)d_out;

  char* ws = (char*)d_ws;
  float*          gi  = (float*)(ws + OFF_GI);
  unsigned short* hsb = (unsigned short*)(ws + OFF_HSB);
  unsigned short* xb  = (unsigned short*)(ws + OFF_XB);
  unsigned short* wob = (unsigned short*)(ws + OFF_WOB);
  unsigned short* wib = (unsigned short*)(ws + OFF_WIB);
  float*          h32 = (float*)(ws + OFF_H32);
  unsigned short* hb  = (unsigned short*)(ws + OFF_HB);
  unsigned*       bar = (unsigned*)(ws + OFF_BAR);

  k_init  <<<64,   256, 0, stream>>>(lh, h32, hb, out + (size_t)BT * V, bar);
  k_cvt   <<<512,  256, 0, stream>>>(w_ih,  wib, G * S);
  k_cvt   <<<2048, 256, 0, stream>>>(w_out, wob, V * S);
  k_gather<<<2048, 256, 0, stream>>>(inp, emb, xb);
  k_gi    <<<dim3(32, 24),  256, 0, stream>>>(xb, wib, b_ih, gi);
  k_gru   <<<32,            192, 0, stream>>>(w_hh, b_hh, gi, h32, hb, hsb, bar);
  k_proj  <<<dim3(64, 125), 256, 0, stream>>>(hsb, wob, b_out, out);
}
